// DyConv_9405978378267
// MI455X (gfx1250) — compile-verified, matrix operand order checked by reading, not run on hardware
//
#include <hip/hip_runtime.h>
#include <math.h>

// ---------------- types ----------------
typedef __attribute__((ext_vector_type(16))) __bf16        bf16x16;
typedef __attribute__((ext_vector_type(8)))  float         f32x8;
typedef __attribute__((ext_vector_type(4)))  unsigned int  u32x4;

union FragU { bf16x16 v; u32x4 u[2]; };

__device__ __forceinline__ unsigned short f2bf(float f) {
  unsigned int u = __float_as_uint(f);
  u += 0x7FFFu + ((u >> 16) & 1u);            // round-to-nearest-even
  return (unsigned short)(u >> 16);
}

// ---------------- problem sizes ----------------
#define NB   16
#define NC   64
#define NH   160
#define NW   160
#define NE   4
#define NR   16
#define HPAD 162
#define WPAD 162
#define KK   576          // Ci * 9
#define MM   256          // E * Co
#define NPIX (NH * NW)    // 25600
#define TILE_N 64         // pixels per block (8 waves: 4 co-tiles x 2 n-halves)

// ---------------- kernel 0: zero padded-x workspace ----------------
__global__ void k_zero(u32x4* p, int n) {
  int i = blockIdx.x * 256 + threadIdx.x;
  if (i < n) { u32x4 z = {}; p[i] = z; }
}

// ---------------- kernel 1: NCHW f32 -> padded NHWC bf16 (LDS transpose) ----------------
__global__ __launch_bounds__(256) void k_pack(const float* __restrict__ x,
                                              unsigned short* __restrict__ xp) {
  __shared__ float tile[64][33];
  int blk = blockIdx.x;                    // 16 * 160 * 5
  int wc = blk % 5;
  int h  = (blk / 5) % 160;
  int b  = blk / 800;
  int t  = threadIdx.x;
  #pragma unroll
  for (int i = 0; i < 8; ++i) {            // coalesced reads along W
    int idx = t + 256 * i;
    int c = idx >> 5, w = idx & 31;
    tile[c][w] = x[((size_t)(b * 64 + c) * 160 + h) * 160 + wc * 32 + w];
  }
  __syncthreads();
  #pragma unroll
  for (int i = 0; i < 8; ++i) {            // coalesced writes along C
    int idx = t + 256 * i;
    int c = idx & 63, w = idx >> 6;
    xp[((size_t)(b * 162 + h + 1) * 162 + (wc * 32 + w + 1)) * 64 + c] = f2bf(tile[c][w]);
  }
}

// ---------------- kernel 2: fold BN into conv weights, pack A[M=256][K=576] bf16 ----------------
__global__ void k_fold(const float* __restrict__ cw, const float* __restrict__ gma,
                       const float* __restrict__ bta, const float* __restrict__ mn,
                       const float* __restrict__ vr, unsigned short* __restrict__ Aout,
                       float* __restrict__ biasOut) {
  int tid = blockIdx.x * 256 + threadIdx.x;
  if (tid >= MM * KK) return;
  int m = tid / KK;                 // m = e*64 + co
  int k = tid - m * KK;             // k = (kh*3+kw)*64 + ci
  int ci = k & 63;
  int kpos = k >> 6;                // kh*3+kw
  float sc = gma[m] * rsqrtf(vr[m] + 1e-3f);
  float wv = cw[(size_t)(m * 64 + ci) * 9 + kpos] * sc;
  Aout[(size_t)m * KK + k] = f2bf(wv);
  if (k == 0) biasOut[m] = bta[m] - mn[m] * sc;
}

// ---------------- kernel 3: global average pool  g[b][c] ----------------
__global__ __launch_bounds__(256) void k_pool(const float* __restrict__ x, float* __restrict__ g) {
  __shared__ float red[256];
  int bc = blockIdx.x;                       // b*64 + c
  const float* p = x + (size_t)bc * NPIX;
  float s = 0.f;
  for (int i = threadIdx.x; i < NPIX; i += 256) s += p[i];
  red[threadIdx.x] = s;
  __syncthreads();
  for (int o = 128; o > 0; o >>= 1) {
    if (threadIdx.x < o) red[threadIdx.x] += red[threadIdx.x + o];
    __syncthreads();
  }
  if (threadIdx.x == 0) g[bc] = red[0] * (1.f / NPIX);
}

// ---------------- kernel 4: router MLP + softmax (1 wave) ----------------
__global__ void k_router(const float* __restrict__ g, const float* __restrict__ fc1,
                         const float* __restrict__ fc2, const float* __restrict__ fc2b,
                         float* __restrict__ wexp) {
  int b = threadIdx.x;
  if (b >= NB) return;
  float hb[NR];
  for (int r = 0; r < NR; ++r) {
    float s = 0.f;
    for (int c = 0; c < NC; ++c) s += g[b * NC + c] * fc1[r * NC + c];
    hb[r] = fmaxf(s, 0.f);
  }
  float lg[NE]; float mx = -1e30f;
  for (int e = 0; e < NE; ++e) {
    float s = fc2b[e];
    for (int r = 0; r < NR; ++r) s += hb[r] * fc2[e * NR + r];
    lg[e] = s; mx = fmaxf(mx, s);
  }
  float den = 0.f;
  for (int e = 0; e < NE; ++e) { lg[e] = __expf(lg[e] - mx); den += lg[e]; }
  float inv = 1.f / den;
  for (int e = 0; e < NE; ++e) wexp[b * NE + e] = lg[e] * inv;
}

// ---------------- kernel 5: fused implicit-GEMM conv + BN bias + SiLU + expert blend ----
// 3-slot rotating register buffers (indices constant after full unroll -> no copies):
// loads for step ks+2 issue before the WMMAs of step ks, so every load clause has
// ~2 full WMMA groups (16 matrix ops) of slack to cover L2 latency.
__global__ __launch_bounds__(256) void k_conv(const unsigned short* __restrict__ xp,
                                              const unsigned short* __restrict__ Ag,
                                              const float* __restrict__ bias,
                                              const float* __restrict__ wr,
                                              float* __restrict__ out) {
  int t = threadIdx.x;
  int wv = t >> 5, lane = t & 31;
  int wn = wv & 1, wm = wv >> 1;            // wave: co-tile [wm*16,+16), n-half wn, all 4 experts
  int lanelo = lane & 15, laneh = lane >> 4;
  int koff = laneh * 8;                     // per-lane-half K sub-offset (bf16 operand layout)

  int pixbase = blockIdx.x * TILE_N;
  int b = pixbase / NPIX;                   // TILE_N divides NPIX -> one image per block
  int pix0 = pixbase - b * NPIX;

  // per-lane pixel base pointers for the two 16-wide n tiles of this wave
  const unsigned short* bn0;
  const unsigned short* bn1;
  {
    int p0 = pix0 + wn * 32 + lanelo;
    int h0 = p0 / 160, w0 = p0 - h0 * 160;
    bn0 = xp + ((size_t)(b * 162 + h0) * 162 + w0) * 64 + koff;
    int p1 = p0 + 16;
    int h1 = p1 / 160, w1 = p1 - h1 * 160;
    bn1 = xp + ((size_t)(b * 162 + h1) * 162 + w1) * 64 + koff;
  }
  const unsigned short* Arow = Ag + (size_t)(wm * 16 + lanelo) * KK + koff;

  float we0 = wr[b * 4 + 0], we1 = wr[b * 4 + 1];
  float we2 = wr[b * 4 + 2], we3 = wr[b * 4 + 3];

  f32x8 acc[4][2];
  #pragma unroll
  for (int e = 0; e < 4; ++e) { f32x8 z = {}; acc[e][0] = z; acc[e][1] = z; }

  // fragment load helpers (kpos/ci0/tap offsets constant-fold after unroll)
#define LOADB(DST, BP, KS)                                                     \
  { int kpos_ = (KS) >> 1; int ci0_ = ((KS) & 1) * 32;                         \
    int off_ = ((kpos_ / 3) * 162 + (kpos_ % 3)) * 64 + ci0_;                  \
    (DST).u[0] = *(const u32x4*)((BP) + off_);                                 \
    (DST).u[1] = *(const u32x4*)((BP) + off_ + 16); }
#define LOADA(DST, E, KS)                                                      \
  { const unsigned short* ap_ = Arow + (size_t)(E) * 64 * KK + (KS) * 32;      \
    (DST).u[0] = *(const u32x4*)(ap_);                                         \
    (DST).u[1] = *(const u32x4*)(ap_ + 16); }
#define LOADSLOT(S, KS)                                                        \
  { LOADB(bufB[S][0], bn0, KS)                                                 \
    LOADB(bufB[S][1], bn1, KS)                                                 \
    LOADA(bufA[S][0], 0, KS)                                                   \
    LOADA(bufA[S][1], 1, KS)                                                   \
    LOADA(bufA[S][2], 2, KS)                                                   \
    LOADA(bufA[S][3], 3, KS) }

  FragU bufA[3][4], bufB[3][2];
  LOADSLOT(0, 0)
  LOADSLOT(1, 1)

  #pragma unroll
  for (int ks = 0; ks < 18; ++ks) {
    if (ks < 16) {                           // prefetch ks+2 into the slot freed at ks-1
      int sn = (ks + 2) % 3;
      LOADSLOT(sn, ks + 2)
    }
    int sc = ks % 3;
    #pragma unroll
    for (int e = 0; e < 4; ++e) {
      acc[e][0] = __builtin_amdgcn_wmma_f32_16x16x32_bf16(false, bufA[sc][e].v, false,
                    bufB[sc][0].v, (short)0, acc[e][0], false, false);
      acc[e][1] = __builtin_amdgcn_wmma_f32_16x16x32_bf16(false, bufA[sc][e].v, false,
                    bufB[sc][1].v, (short)0, acc[e][1], false, false);
    }
  }
#undef LOADSLOT
#undef LOADA
#undef LOADB

  // ---- epilogue: bias + SiLU per expert, router-weighted blend, single store ----
  #pragma unroll
  for (int tt = 0; tt < 2; ++tt) {
    int colg = pix0 + wn * 32 + tt * 16 + lanelo;
    #pragma unroll
    for (int vi = 0; vi < 8; ++vi) {
      int co = wm * 16 + vi + laneh * 8;     // C layout: rows 0-7 lanes 0-15, rows 8-15 lanes 16-31
      float y0 = acc[0][tt][vi] + bias[0 * 64 + co];
      float y1 = acc[1][tt][vi] + bias[1 * 64 + co];
      float y2 = acc[2][tt][vi] + bias[2 * 64 + co];
      float y3 = acc[3][tt][vi] + bias[3 * 64 + co];
      float s = we0 * (y0 / (1.f + __expf(-y0)))
              + we1 * (y1 / (1.f + __expf(-y1)))
              + we2 * (y2 / (1.f + __expf(-y2)))
              + we3 * (y3 / (1.f + __expf(-y3)));
      out[(size_t)(b * 64 + co) * NPIX + colg] = s;
    }
  }
}

// ---------------- host launcher ----------------
extern "C" void kernel_launch(void* const* d_in, const int* in_sizes, int n_in,
                              void* d_out, int out_size, void* d_ws, size_t ws_size,
                              hipStream_t stream) {
  const float* x    = (const float*)d_in[0];
  const float* fc1  = (const float*)d_in[1];
  const float* fc2  = (const float*)d_in[2];
  const float* fc2b = (const float*)d_in[3];
  const float* cw   = (const float*)d_in[4];
  const float* gma  = (const float*)d_in[5];
  const float* bta  = (const float*)d_in[6];
  const float* mn   = (const float*)d_in[7];
  const float* vr   = (const float*)d_in[8];
  float* out = (float*)d_out;

  char* ws = (char*)d_ws;
  const size_t XPAD_ELEMS = (size_t)NB * HPAD * WPAD * NC;        // 26,873,856 bf16
  unsigned short* xpad = (unsigned short*)ws;
  unsigned short* Af   = (unsigned short*)(ws + XPAD_ELEMS * 2);  // 16B-aligned
  float* biasp = (float*)(ws + XPAD_ELEMS * 2 + (size_t)MM * KK * 2);
  float* g     = biasp + MM;
  float* wexp  = g + NB * NC;

  int nzero = (int)(XPAD_ELEMS * 2 / 16);
  k_zero  <<<(nzero + 255) / 256, 256, 0, stream>>>((u32x4*)xpad, nzero);
  k_pack  <<<NB * NH * 5, 256, 0, stream>>>(x, xpad);
  k_fold  <<<(MM * KK + 255) / 256, 256, 0, stream>>>(cw, gma, bta, mn, vr, Af, biasp);
  k_pool  <<<NB * NC, 256, 0, stream>>>(x, g);
  k_router<<<1, 32, 0, stream>>>(g, fc1, fc2, fc2b, wexp);
  k_conv  <<<(NB * NPIX) / TILE_N, 256, 0, stream>>>(xpad, Af, biasp, wexp, out);
}
